// GAT_3_6408091205943
// MI455X (gfx1250) — matrix operand order repaired and checked by reading, not compiled, not run on hardware
//
#include <hip/hip_runtime.h>
#include <hip/hip_bf16.h>

// GAT-3 pipeline for MI455X (gfx1250, wave32).
// Dense projections: v_wmma_f32_16x16x32_f16 with f16 operands staged so every
// fragment is loaded with unconditional global_load_b128 (no per-element guards).
// Edge softmax/aggregation, BN, pooling: fp32 memory/atomic-bound passes.

#define N_NODES 50000
#define N_EDGES 400000
#define F_IN    128
#define N_GRAPH 64
#define N_CLS   10

typedef __attribute__((ext_vector_type(16))) _Float16 v16h;
typedef __attribute__((ext_vector_type(8)))  _Float16 v8h;
typedef __attribute__((ext_vector_type(8)))  float    v8f;

// ---------------------------------------------------------------- utilities

__device__ __forceinline__ void atomicMaxFloat(float* addr, float val) {
    // sign-split trick: float ordering == int ordering for >=0,
    // == reversed unsigned ordering for <0.
    if (val >= 0.0f) atomicMax((int*)addr, __float_as_int(val));
    else             atomicMin((unsigned int*)addr, (unsigned int)__float_as_int(val));
}

__global__ void fill_const_kernel(float* __restrict__ p, float v, int n) {
    int i = blockIdx.x * blockDim.x + threadIdx.x;
    if (i < n) p[i] = v;
}

__global__ void fill_bias_kernel(float* __restrict__ out, const float* __restrict__ bias,
                                 int n, int C) {
    int i = blockIdx.x * blockDim.x + threadIdx.x;
    if (i < n * C) out[i] = bias[i % C];
}

// f32 [n,C] -> f16 [n,Cpad], zero padded columns C..Cpad-1
__global__ void row_f32_to_f16_kernel(const float* __restrict__ in,
                                      _Float16* __restrict__ out,
                                      int n, int C, int Cpad) {
    int i = blockIdx.x * blockDim.x + threadIdx.x;
    if (i >= n * Cpad) return;
    int r = i / Cpad, c = i % Cpad;
    out[i] = (c < C) ? (_Float16)in[(size_t)r * C + c] : (_Float16)0.0f;
}

// weight f32 [K,Nc] -> f16 transposed [Ncpad,Kpad], zero padded
__global__ void wt_transpose_f16_kernel(const float* __restrict__ in,
                                        _Float16* __restrict__ out,
                                        int K, int Nc, int Kpad, int Ncpad) {
    int i = blockIdx.x * blockDim.x + threadIdx.x;
    if (i >= Ncpad * Kpad) return;
    int r = i / Kpad, k = i % Kpad;   // r = output col of original B
    out[i] = (r < Nc && k < K) ? (_Float16)in[(size_t)k * Nc + r] : (_Float16)0.0f;
}

// ------------------------------------------------- WMMA tiled GEMM + bias
// C[M,NcStore] = A16[M,K] @ B (given as BT16[Ncpad,K]) + bias
// Requirements (guaranteed by call sites): M%16==0, K%32==0, Ncpad%16==0.
// One wave per 16x16 C tile. Fragment k order used here (contiguous, same for A and B):
//  - A 16x32 f16: lane half 0 holds K 0..15, half 1 holds K 16..31
//    -> two contiguous 8-half runs from row-major A16 => 2x b128 loads.
//  - B 32x16 f16: lane half 0 holds K 0..15, half 1 holds K 16..31
//    -> one contiguous 16-half run from BT16 => 2x b128 loads.
__global__ void wmma_gemm_f16_kernel(const _Float16* __restrict__ A16,
                                     const _Float16* __restrict__ BT16,
                                     const float* __restrict__ bias,
                                     float* __restrict__ C,
                                     int M, int K, int Ncpad, int NcStore) {
    int lane   = threadIdx.x & 31;
    int waveId = threadIdx.x >> 5;
    int tilesN = Ncpad >> 4;
    int tilesM = M >> 4;
    int tile   = blockIdx.x * (blockDim.x >> 5) + waveId;
    if (tile >= tilesM * tilesN) return;
    int tm = tile / tilesN, tn = tile % tilesN;
    int row0 = tm << 4, col0 = tn << 4;

    int half = lane >> 4;          // 0: lanes 0-15, 1: lanes 16-31
    int lid  = lane & 15;
    const _Float16* aRow = A16  + (size_t)(row0 + lid) * K + half * 16;  // A row
    const _Float16* bRow = BT16 + (size_t)(col0 + lid) * K + half * 16;  // B col

    v8f acc = {};
    for (int k0 = 0; k0 < K; k0 += 32) {
        v8h alo = *(const v8h*)(aRow + k0);        // K: kbA+0..7
        v8h ahi = *(const v8h*)(aRow + k0 + 8);    // K: kbA+8..15
        v8h blo = *(const v8h*)(bRow + k0);        // K: kbB+0..7
        v8h bhi = *(const v8h*)(bRow + k0 + 8);    // K: kbB+8..15
        v16h a = __builtin_shufflevector(alo, ahi, 0,1,2,3,4,5,6,7,
                                                   8,9,10,11,12,13,14,15);
        v16h b = __builtin_shufflevector(blo, bhi, 0,1,2,3,4,5,6,7,
                                                   8,9,10,11,12,13,14,15);
        acc = __builtin_amdgcn_wmma_f32_16x16x32_f16(
            false, a, false, b, (short)0, acc, false, false);
    }
    // C/D 16x16 f32: lane half selects M base (0 or 8), VGPR v = M offset.
    int bn = col0 + lid;
    if (bn < NcStore) {
        float bs = bias ? bias[bn] : 0.0f;
        #pragma unroll
        for (int v = 0; v < 8; ++v) {
            int rm = row0 + half * 8 + v;
            C[(size_t)rm * NcStore + bn] = acc[v] + bs;
        }
    }
}

// ------------------------------------------------------------ GAT edge phase

__global__ void node_scores_kernel(const float* __restrict__ h,
                                   const float* __restrict__ a_src,
                                   const float* __restrict__ a_dst,
                                   float* __restrict__ ss, float* __restrict__ sd,
                                   int n, int H, int C) {
    int i = blockIdx.x * blockDim.x + threadIdx.x;
    if (i >= n * H) return;
    int node = i / H, hh = i % H;
    const float* hp = h + (size_t)node * H * C + (size_t)hh * C;
    const float* ap = a_src + hh * C;
    const float* dp = a_dst + hh * C;
    float s1 = 0.0f, s2 = 0.0f;
    for (int c = 0; c < C; ++c) { float v = hp[c]; s1 += v * ap[c]; s2 += v * dp[c]; }
    ss[i] = s1;
    sd[i] = s2;
}

__device__ __forceinline__ void edge_endpoints(const int* __restrict__ ei,
                                               int e, int E, int& s, int& d) {
    // edge_index is [2,E] row-major; edges >= E are self loops (e-E, e-E).
    if (e < E) { s = ei[e]; d = ei[E + e]; }
    else       { s = e - E; d = e - E; }
}

__global__ void edge_max_kernel(const int* __restrict__ ei, int E, int n, int H,
                                const float* __restrict__ ss,
                                const float* __restrict__ sd,
                                float* __restrict__ m) {
    int i = blockIdx.x * blockDim.x + threadIdx.x;
    int tot = (E + n) * H;
    if (i >= tot) return;
    int e = i / H, hh = i % H;
    int s, d; edge_endpoints(ei, e, E, s, d);
    float sc = ss[(size_t)s * H + hh] + sd[(size_t)d * H + hh];
    sc = sc > 0.0f ? sc : 0.2f * sc;                 // leaky_relu(0.2)
    atomicMaxFloat(&m[(size_t)d * H + hh], sc);
}

__global__ void edge_sum_kernel(const int* __restrict__ ei, int E, int n, int H,
                                const float* __restrict__ ss,
                                const float* __restrict__ sd,
                                const float* __restrict__ m,
                                float* __restrict__ z) {
    int i = blockIdx.x * blockDim.x + threadIdx.x;
    int tot = (E + n) * H;
    if (i >= tot) return;
    int e = i / H, hh = i % H;
    int s, d; edge_endpoints(ei, e, E, s, d);
    size_t di = (size_t)d * H + hh;
    float sc = ss[(size_t)s * H + hh] + sd[di];
    sc = sc > 0.0f ? sc : 0.2f * sc;
    atomicAdd(&z[di], expf(sc - m[di]));
}

__global__ void edge_scatter_kernel(const int* __restrict__ ei, int E, int n,
                                    int H, int C,
                                    const float* __restrict__ ss,
                                    const float* __restrict__ sd,
                                    const float* __restrict__ m,
                                    const float* __restrict__ z,
                                    const float* __restrict__ h_in,
                                    float* __restrict__ out) {
    int i = blockIdx.x * blockDim.x + threadIdx.x;
    int tot = (E + n) * H;
    if (i >= tot) return;
    int e = i / H, hh = i % H;
    int s, d; edge_endpoints(ei, e, E, s, d);
    size_t di = (size_t)d * H + hh;
    float sc = ss[(size_t)s * H + hh] + sd[di];
    sc = sc > 0.0f ? sc : 0.2f * sc;
    float alpha = expf(sc - m[di]) / (z[di] + 1e-16f);
    const float* hp = h_in + (size_t)s * H * C + (size_t)hh * C;
    float*       op = out  + (size_t)d * H * C + (size_t)hh * C;
    for (int c = 0; c < C; ++c) atomicAdd(&op[c], hp[c] * alpha);
}

// ------------------------------------------------------------ BatchNorm + ReLU

__global__ void bn_stats_kernel(const float* __restrict__ x, int n, int C,
                                float* __restrict__ sum, float* __restrict__ sq) {
    const int RPB = 64;                 // rows per block
    int r0 = blockIdx.x * RPB;
    int r1 = min(r0 + RPB, n);
    float acc[2] = {0.0f, 0.0f}, accq[2] = {0.0f, 0.0f};   // C <= 512, blockDim=256
    for (int r = r0; r < r1; ++r) {
        int j = 0;
        for (int c = threadIdx.x; c < C; c += blockDim.x, ++j) {
            float v = x[(size_t)r * C + c];
            acc[j] += v; accq[j] += v * v;
        }
    }
    int j = 0;
    for (int c = threadIdx.x; c < C; c += blockDim.x, ++j) {
        atomicAdd(&sum[c], acc[j]);
        atomicAdd(&sq[c],  accq[j]);
    }
}

__global__ void bn_finalize_kernel(const float* __restrict__ sum,
                                   const float* __restrict__ sq,
                                   const float* __restrict__ g,
                                   const float* __restrict__ be,
                                   float* __restrict__ scale,
                                   float* __restrict__ shift, int n, int C) {
    int c = threadIdx.x;
    if (c >= C) return;
    float inv_n = 1.0f / (float)n;
    float mean  = sum[c] * inv_n;
    float var   = sq[c] * inv_n - mean * mean;
    float sc    = g[c] * rsqrtf(var + 1e-5f);
    scale[c] = sc;
    shift[c] = be[c] - mean * sc;
}

// y = relu(x*scale+shift), written back to x (f32); optionally also writes a
// zero-padded f16 copy [n,Cpad] used as the next GEMM's A operand.
__global__ void bn_apply_relu_kernel(float* __restrict__ x,
                                     const float* __restrict__ scale,
                                     const float* __restrict__ shift,
                                     int n, int C,
                                     _Float16* __restrict__ out16, int Cpad) {
    int i = blockIdx.x * blockDim.x + threadIdx.x;
    if (i >= n * Cpad) return;
    int r = i / Cpad, c = i % Cpad;
    if (c < C) {
        size_t xi = (size_t)r * C + c;
        float v = x[xi] * scale[c] + shift[c];
        v = v > 0.0f ? v : 0.0f;
        x[xi] = v;
        if (out16) out16[i] = (_Float16)v;
    } else if (out16) {
        out16[i] = (_Float16)0.0f;
    }
}

// ------------------------------------------------------------ mean pool

__global__ void pool_add_kernel(const float* __restrict__ h,
                                const int* __restrict__ batch,
                                float* __restrict__ gsum, float* __restrict__ gcnt,
                                int n, int C) {
    int i = blockIdx.x * blockDim.x + threadIdx.x;
    if (i >= n * C) return;
    int node = i / C, c = i % C;
    int g = batch[node];
    atomicAdd(&gsum[(size_t)g * C + c], h[i]);
    if (c == 0) atomicAdd(&gcnt[g], 1.0f);
}

__global__ void pool_div_kernel(const float* __restrict__ gsum,
                                const float* __restrict__ gcnt,
                                float* __restrict__ pooled, int G, int C) {
    int i = blockIdx.x * blockDim.x + threadIdx.x;
    if (i >= G * C) return;
    int g = i / C;
    pooled[i] = gsum[i] / fmaxf(gcnt[g], 1.0f);
}

// ------------------------------------------------------------ host driver

static inline int ceil_div(int a, int b) { return (a + b - 1) / b; }

static void launch_gemm(const _Float16* A16, const _Float16* BT16, const float* bias,
                        float* C, int M, int K, int Ncpad, int NcStore,
                        hipStream_t stream) {
    int tiles  = (M >> 4) * (Ncpad >> 4);
    int blocks = ceil_div(tiles, 4);        // 4 waves (tiles) per 128-thread block
    wmma_gemm_f16_kernel<<<blocks, 128, 0, stream>>>(A16, BT16, bias, C,
                                                     M, K, Ncpad, NcStore);
}

static void launch_fill(float* p, float v, int n, hipStream_t stream) {
    fill_const_kernel<<<ceil_div(n, 256), 256, 0, stream>>>(p, v, n);
}

static void launch_wt(const float* w, _Float16* wt, int K, int Nc, int Kpad, int Ncpad,
                      hipStream_t stream) {
    wt_transpose_f16_kernel<<<ceil_div(Ncpad * Kpad, 256), 256, 0, stream>>>(
        w, wt, K, Nc, Kpad, Ncpad);
}

static void gat_edge_phase(const float* h_in, const int* ei,
                           const float* a_src, const float* a_dst, const float* bias,
                           float* out, float* ss, float* sd, float* m, float* z,
                           int n, int E, int H, int C, hipStream_t stream) {
    node_scores_kernel<<<ceil_div(n * H, 256), 256, 0, stream>>>(h_in, a_src, a_dst,
                                                                 ss, sd, n, H, C);
    launch_fill(m, -3.0e38f, n * H, stream);
    launch_fill(z, 0.0f,     n * H, stream);
    int etot = (E + n) * H;
    edge_max_kernel<<<ceil_div(etot, 256), 256, 0, stream>>>(ei, E, n, H, ss, sd, m);
    edge_sum_kernel<<<ceil_div(etot, 256), 256, 0, stream>>>(ei, E, n, H, ss, sd, m, z);
    fill_bias_kernel<<<ceil_div(n * H * C, 256), 256, 0, stream>>>(out, bias, n, H * C);
    edge_scatter_kernel<<<ceil_div(etot, 256), 256, 0, stream>>>(ei, E, n, H, C,
                                                                 ss, sd, m, z, h_in, out);
}

static void batch_norm_relu(float* x, const float* g, const float* be,
                            float* csum, float* csq, float* scale, float* shift,
                            int n, int C, _Float16* out16, int Cpad,
                            hipStream_t stream) {
    launch_fill(csum, 0.0f, C, stream);
    launch_fill(csq,  0.0f, C, stream);
    bn_stats_kernel<<<ceil_div(n, 64), 256, 0, stream>>>(x, n, C, csum, csq);
    bn_finalize_kernel<<<1, C, 0, stream>>>(csum, csq, g, be, scale, shift, n, C);
    bn_apply_relu_kernel<<<ceil_div(n * Cpad, 256), 256, 0, stream>>>(
        x, scale, shift, n, C, out16, Cpad);
}

extern "C" void kernel_launch(void* const* d_in, const int* in_sizes, int n_in,
                              void* d_out, int out_size, void* d_ws, size_t ws_size,
                              hipStream_t stream) {
    const float* x   = (const float*)d_in[0];
    const int*   ei  = (const int*)  d_in[1];
    const int*   bat = (const int*)  d_in[2];
    const float *W1  = (const float*)d_in[3],  *as1 = (const float*)d_in[4],
                *ad1 = (const float*)d_in[5],  *b1  = (const float*)d_in[6],
                *g1  = (const float*)d_in[7],  *be1 = (const float*)d_in[8],
                *lw1 = (const float*)d_in[9],  *lb1 = (const float*)d_in[10],
                *gl1 = (const float*)d_in[11], *bel1= (const float*)d_in[12],
                *W2  = (const float*)d_in[13], *as2 = (const float*)d_in[14],
                *ad2 = (const float*)d_in[15], *b2  = (const float*)d_in[16],
                *g2  = (const float*)d_in[17], *be2 = (const float*)d_in[18],
                *lw2 = (const float*)d_in[19], *lb2 = (const float*)d_in[20],
                *gl2 = (const float*)d_in[21], *bel2= (const float*)d_in[22],
                *W3  = (const float*)d_in[23], *as3 = (const float*)d_in[24],
                *ad3 = (const float*)d_in[25], *b3  = (const float*)d_in[26],
                *g3  = (const float*)d_in[27], *be3 = (const float*)d_in[28],
                *lwf = (const float*)d_in[29], *lbf = (const float*)d_in[30];
    float* out = (float*)d_out;

    const int n = N_NODES, E = N_EDGES, G = N_GRAPH;

    // workspace carve-up (256B aligned slices)
    size_t off = 0;
    char* base = (char*)d_ws;
    auto alloc = [&](size_t bytes) -> void* {
        void* p = base + off;
        off += (bytes + 255) & ~(size_t)255;
        return p;
    };
    float*     bufA  = (float*)alloc((size_t)n * 512 * 4);
    float*     bufB  = (float*)alloc((size_t)n * 512 * 4);
    _Float16*  h16   = (_Float16*)alloc((size_t)n * 512 * 2);  // staged A operand
    float*     ss    = (float*)alloc((size_t)n * 20 * 4);
    float*     sd    = (float*)alloc((size_t)n * 20 * 4);
    float*     mmax  = (float*)alloc((size_t)n * 20 * 4);
    float*     zsum  = (float*)alloc((size_t)n * 20 * 4);
    float*     csum  = (float*)alloc(512 * 4);
    float*     csq   = (float*)alloc(512 * 4);
    float*     scale = (float*)alloc(512 * 4);
    float*     shift = (float*)alloc(512 * 4);
    float*     gsum  = (float*)alloc((size_t)G * 384 * 4);
    float*     gcnt  = (float*)alloc((size_t)G * 4);
    float*     pool  = (float*)alloc((size_t)G * 384 * 4);
    _Float16*  w1t   = (_Float16*)alloc((size_t)320 * 128 * 2);
    _Float16*  lw1t  = (_Float16*)alloc((size_t)16  * 320 * 2);
    _Float16*  w2t   = (_Float16*)alloc((size_t)512 * 32  * 2);
    _Float16*  lw2t  = (_Float16*)alloc((size_t)32  * 512 * 2);
    _Float16*  w3t   = (_Float16*)alloc((size_t)384 * 32  * 2);
    _Float16*  lwft  = (_Float16*)alloc((size_t)16  * 384 * 2);

    // ---- stage weights as f16 B^T (padded to fragment-friendly shapes) ----
    launch_wt(W1,  w1t,  128, 320, 128, 320, stream);
    launch_wt(lw1, lw1t, 320, 16,  320, 16,  stream);
    launch_wt(W2,  w2t,  16,  512, 32,  512, stream);   // K 16 -> padded 32
    launch_wt(lw2, lw2t, 512, 32,  512, 32,  stream);
    launch_wt(W3,  w3t,  32,  384, 32,  384, stream);
    launch_wt(lwf, lwft, 384, 10,  384, 16,  stream);   // Nc 10 -> padded 16

    // ---- layer 1: GATConv(128 -> 16, H=20) ----
    row_f32_to_f16_kernel<<<ceil_div(n * F_IN, 256), 256, 0, stream>>>(x, h16,
                                                                       n, F_IN, F_IN);
    launch_gemm(h16, w1t, nullptr, bufA, n, F_IN, 320, 320, stream);   // h = x @ W1
    gat_edge_phase(bufA, ei, as1, ad1, b1, bufB, ss, sd, mmax, zsum,
                   n, E, 20, 16, stream);
    batch_norm_relu(bufB, g1, be1, csum, csq, scale, shift, n, 320, h16, 320, stream);
    launch_gemm(h16, lw1t, lb1, bufA, n, 320, 16, 16, stream);         // linear 320->16
    batch_norm_relu(bufA, gl1, bel1, csum, csq, scale, shift, n, 16, h16, 32, stream);

    // ---- layer 2: GATConv(16 -> 32, H=16) ----
    launch_gemm(h16, w2t, nullptr, bufB, n, 32, 512, 512, stream);     // K padded to 32
    gat_edge_phase(bufB, ei, as2, ad2, b2, bufA, ss, sd, mmax, zsum,
                   n, E, 16, 32, stream);
    batch_norm_relu(bufA, g2, be2, csum, csq, scale, shift, n, 512, h16, 512, stream);
    launch_gemm(h16, lw2t, lb2, bufB, n, 512, 32, 32, stream);         // linear 512->32
    batch_norm_relu(bufB, gl2, bel2, csum, csq, scale, shift, n, 32, h16, 32, stream);

    // ---- layer 3: GATConv(32 -> 48, H=8) ----
    launch_gemm(h16, w3t, nullptr, bufA, n, 32, 384, 384, stream);
    gat_edge_phase(bufA, ei, as3, ad3, b3, bufB, ss, sd, mmax, zsum,
                   n, E, 8, 48, stream);
    batch_norm_relu(bufB, g3, be3, csum, csq, scale, shift, n, 384, nullptr, 384, stream);

    // ---- global mean pool + classifier ----
    launch_fill(gsum, 0.0f, G * 384, stream);
    launch_fill(gcnt, 0.0f, G, stream);
    pool_add_kernel<<<ceil_div(n * 384, 256), 256, 0, stream>>>(bufB, bat, gsum, gcnt,
                                                                n, 384);
    pool_div_kernel<<<ceil_div(G * 384, 256), 256, 0, stream>>>(gsum, gcnt, pool, G, 384);
    row_f32_to_f16_kernel<<<ceil_div(G * 384, 256), 256, 0, stream>>>(pool, h16,
                                                                      G, 384, 384);
    launch_gemm(h16, lwft, lbf, out, G, 384, 16, N_CLS, stream);       // [64,10]
}
